// StyleLossPass1_1640677507468
// MI455X (gfx1250) — compile-verified, matrix operand order checked by reading, not run on hardware
//
#include <hip/hip_runtime.h>
#include <hip/hip_bf16.h>

// ---------------------------------------------------------------------------
// StyleLoss for MI455X (gfx1250, wave32, WMMA + async vmem->LDS staging).
// scores GEMM (4096x4096x4608) + grams (512x512x36864 x2) on the bf16
// v_wmma_f32_16x16x32_bf16 path, fp32 accum. Operands staged cooperatively
// through LDS with GLOBAL_LOAD_ASYNC_TO_LDS_B128 (ASYNCcnt), double-buffered.
// Argmax fused into GEMM epilogue via shuffle-reduce + packed u64 atomicMax.
// Deterministic split-K for the grams (partial buffers, no fp atomics).
// ---------------------------------------------------------------------------

typedef __attribute__((ext_vector_type(8)))  __bf16 v8bf;
typedef __attribute__((ext_vector_type(16))) __bf16 v16bf;
typedef __attribute__((ext_vector_type(8)))  float  v8f;

#define C_CH   512
#define IMG_H  192
#define HW     36864         // 192*192
#define NPATCH 4096          // 64*64 non-overlapping 3x3 patches
#define KPAT   4608          // 512*9
#define GRAM_N 512
#define KSPLIT 8
#define GRAM_KCHUNK (HW / KSPLIT)   // 4608, multiple of 32

// Workgroup GEMM tile: 8 waves (2x4), each wave 32x64 -> 64x256 per block.
#define TILE_M   64
#define TILE_N   256
#define LDS_ROWB 80          // bytes per staged 32-element bf16 row (64B + 16B pad)
#define LDS_ROW  40          // same, in bf16 elements
#define A_STAGE_ELEMS (TILE_M * LDS_ROW)    // 2560
#define B_STAGE_ELEMS (TILE_N * LDS_ROW)    // 10240
#define A_STAGE_BYTES (A_STAGE_ELEMS * 2)   // 5120
#define B_STAGE_BYTES (B_STAGE_ELEMS * 2)   // 20480

// ---------------- stage 1: pack fp32 feature maps -> bf16 patch matrix -----
// out[p*KPAT + k], p = ph*64+pw, k = c*9 + dy*3 + dx  (matches jax _patches)
__global__ void pack_patches(const float* __restrict__ fm,
                             __hip_bfloat16* __restrict__ out) {
  size_t i = (size_t)blockIdx.x * blockDim.x + threadIdx.x;
  if (i >= (size_t)NPATCH * KPAT) return;
  int p = (int)(i / KPAT), k = (int)(i % KPAT);
  int c = k / 9, r = k % 9, dy = r / 3, dx = r % 3;
  int ph = p >> 6, pw = p & 63;
  float v = fm[(size_t)c * HW + (size_t)(ph * 3 + dy) * IMG_H + (pw * 3 + dx)];
  out[i] = __float2bfloat16(v);
}

// ---------------- async copy of one 32-K stage into LDS --------------------
// 256 threads move A tile (64 rows x 64B) + B tile (256 rows x 64B) as 16B
// chunks: thread t -> A row t>>2 chunk t&3, plus 4 B rows (t>>2)+j*64.
__device__ __forceinline__ void async_stage(const __hip_bfloat16* __restrict__ Acol,
                                            const __hip_bfloat16* __restrict__ Bcol,
                                            unsigned ldsA, unsigned ldsB,
                                            int K, int t) {
  int row = t >> 2, c = t & 3;
  {
    unsigned long long ga =
        (unsigned long long)(size_t)(Acol + (size_t)row * K + c * 8);
    unsigned la = ldsA + (unsigned)(row * LDS_ROWB + c * 16);
    asm volatile("global_load_async_to_lds_b128 %0, %1, off"
                 :: "v"(la), "v"(ga) : "memory");
  }
#pragma unroll
  for (int j = 0; j < 4; ++j) {
    int rb = row + j * 64;
    unsigned long long gb =
        (unsigned long long)(size_t)(Bcol + (size_t)rb * K + c * 8);
    unsigned lb = ldsB + (unsigned)(rb * LDS_ROWB + c * 16);
    asm volatile("global_load_async_to_lds_b128 %0, %1, off"
                 :: "v"(lb), "v"(gb) : "memory");
  }
}

// ---------------- fragment load from LDS (ISA 16-bit A 16x32 layout) -------
// lane&15 = matrix row; lane>>4 selects K groups {0..7,16..23} / {8..15,24..31}
__device__ __forceinline__ v16bf lds_frag(const __hip_bfloat16* buf, int row,
                                          int lane) {
  const __hip_bfloat16* rp = buf + row * LDS_ROW + ((lane >> 4) << 3);
  v8bf lo = *reinterpret_cast<const v8bf*>(rp);        // K = kk + half*8 ..
  v8bf hi = *reinterpret_cast<const v8bf*>(rp + 16);   // K = kk+16+half*8 ..
  return __builtin_shufflevector(lo, hi, 0, 1, 2, 3, 4, 5, 6, 7,
                                 8, 9, 10, 11, 12, 13, 14, 15);
}

__device__ __forceinline__ v8f wmma_bf16(v16bf a, v16bf b, v8f c) {
  return __builtin_amdgcn_wmma_f32_16x16x32_bf16(
      /*neg_a=*/false, a, /*neg_b=*/false, b,
      /*c_mod=*/(short)0, c, /*reuse_a=*/false, /*reuse_b=*/false);
}

// ---------------- shared LDS-pipelined 64x256 workgroup GEMM body ----------
// Accumulates this wave's 32x64 sub-tile over K range [kk0, kk1).
__device__ __forceinline__ void gemm_tile_pipeline(
    const __hip_bfloat16* __restrict__ A,   // [M,K] bf16 row-major
    const __hip_bfloat16* __restrict__ B,   // [N,K] bf16 row-major
    int K, int kk0, int kk1, int m0g, int n0g,
    __hip_bfloat16* shA, __hip_bfloat16* shB,   // [2][stage]
    v8f acc[2][4]) {
  int t = (int)threadIdx.x;
  int lane = t & 31;
  int r = lane & 15;
  int waveId = t >> 5;
  int m0w = (waveId >> 2) * 32;   // wave row offset in workgroup tile
  int n0w = (waveId & 3) * 64;    // wave col offset in workgroup tile

  const __hip_bfloat16* Abase = A + (size_t)m0g * K;
  const __hip_bfloat16* Bbase = B + (size_t)n0g * K;
  unsigned ldsA0 = (unsigned)(size_t)shA;
  unsigned ldsB0 = (unsigned)(size_t)shB;

  int steps = (kk1 - kk0) >> 5;
  async_stage(Abase + kk0, Bbase + kk0, ldsA0, ldsB0, K, t);   // stage 0

  for (int s = 0; s < steps; ++s) {
    if (s + 1 < steps) {
      int kn = kk0 + (s + 1) * 32;
      async_stage(Abase + kn, Bbase + kn,
                  ldsA0 + (unsigned)((s + 1) & 1) * A_STAGE_BYTES,
                  ldsB0 + (unsigned)((s + 1) & 1) * B_STAGE_BYTES, K, t);
      // older stage's 5 loads done; the 5 just issued may stay in flight
      asm volatile("s_wait_asynccnt 0x5" ::: "memory");
    } else {
      asm volatile("s_wait_asynccnt 0x0" ::: "memory");
    }
    __syncthreads();   // stage s visible to all waves

    const __hip_bfloat16* bufA = shA + (s & 1) * A_STAGE_ELEMS;
    const __hip_bfloat16* bufB = shB + (s & 1) * B_STAGE_ELEMS;
    v16bf a0 = lds_frag(bufA, m0w + r, lane);
    v16bf a1 = lds_frag(bufA, m0w + 16 + r, lane);
    v16bf b0 = lds_frag(bufB, n0w + r, lane);
    v16bf b1 = lds_frag(bufB, n0w + 16 + r, lane);
    v16bf b2 = lds_frag(bufB, n0w + 32 + r, lane);
    v16bf b3 = lds_frag(bufB, n0w + 48 + r, lane);
    acc[0][0] = wmma_bf16(a0, b0, acc[0][0]);
    acc[0][1] = wmma_bf16(a0, b1, acc[0][1]);
    acc[0][2] = wmma_bf16(a0, b2, acc[0][2]);
    acc[0][3] = wmma_bf16(a0, b3, acc[0][3]);
    acc[1][0] = wmma_bf16(a1, b0, acc[1][0]);
    acc[1][1] = wmma_bf16(a1, b1, acc[1][1]);
    acc[1][2] = wmma_bf16(a1, b2, acc[1][2]);
    acc[1][3] = wmma_bf16(a1, b3, acc[1][3]);
    __syncthreads();   // all waves done reading buf (s&1) before reuse
  }
}

// ---------------- stage 2: scores = IP @ SP^T, fused row-argmax ------------
__global__ void __launch_bounds__(256)
gemm_nt_argmax(const __hip_bfloat16* __restrict__ A,   // [M,K]
               const __hip_bfloat16* __restrict__ B,   // [N,K]
               unsigned long long* __restrict__ packed,
               int M, int N, int K) {
  __shared__ __hip_bfloat16 shA[2 * A_STAGE_ELEMS];
  __shared__ __hip_bfloat16 shB[2 * B_STAGE_ELEMS];
  int tilesN = N / TILE_N;
  int m0g = ((int)blockIdx.x / tilesN) * TILE_M;
  int n0g = ((int)blockIdx.x % tilesN) * TILE_N;

  v8f acc[2][4] = {};
  gemm_tile_pipeline(A, B, K, 0, K, m0g, n0g, shA, shB, acc);

  int lane = (int)threadIdx.x & 31;
  int r = lane & 15, half = lane >> 4;
  int waveId = (int)threadIdx.x >> 5;
  int m0 = m0g + (waveId >> 2) * 32;
  int n0 = n0g + (waveId & 3) * 64;

  // C/D layout: VGPR j of lanes 0-15 = row j, lanes 16-31 = row j+8; col=lane&15
  for (int ms = 0; ms < 2; ++ms) {
    for (int j = 0; j < 8; ++j) {
      float best = acc[ms][0][j];
      unsigned bn = (unsigned)(n0 + r);
      for (int ns = 1; ns < 4; ++ns) {
        float v = acc[ms][ns][j];
        unsigned nn = (unsigned)(n0 + ns * 16 + r);
        if (v > best || (v == best && nn < bn)) { best = v; bn = nn; }
      }
      for (int off = 8; off >= 1; off >>= 1) {   // xor stays within 16-lane half
        float ov = __shfl_xor(best, off, 32);
        unsigned on = (unsigned)__shfl_xor((int)bn, off, 32);
        if (ov > best || (ov == best && on < bn)) { best = ov; bn = on; }
      }
      if (r == 0) {
        int row = m0 + ms * 16 + j + (half << 3);
        unsigned ub = __float_as_uint(best);
        ub = (ub & 0x80000000u) ? ~ub : (ub | 0x80000000u);   // sortable key
        unsigned long long key =
            ((unsigned long long)ub << 32) | (unsigned)(~bn);  // ~n: tie->min n
        atomicMax(&packed[row], key);
      }
    }
  }
}

__global__ void decode_idx(const unsigned long long* __restrict__ packed,
                           unsigned* __restrict__ idx) {
  int i = blockIdx.x * blockDim.x + threadIdx.x;
  if (i < NPATCH) idx[i] = ~(unsigned)(packed[i] & 0xFFFFFFFFull);
}

// ---------------- stage 3: masked Gram operands (bf16, row-major [C,HW]) ---
__global__ void pack_masked(const float* __restrict__ x,
                            const float* __restrict__ mask,
                            __hip_bfloat16* __restrict__ out) {
  size_t i = (size_t)blockIdx.x * blockDim.x + threadIdx.x;
  if (i >= (size_t)C_CH * HW) return;
  int hw = (int)(i % HW);
  out[i] = __float2bfloat16(x[i] * mask[hw]);
}

__global__ void pack_matched(const float* __restrict__ style,
                             const float* __restrict__ mask,
                             const unsigned* __restrict__ idx,
                             __hip_bfloat16* __restrict__ out) {
  size_t i = (size_t)blockIdx.x * blockDim.x + threadIdx.x;
  if (i >= (size_t)C_CH * HW) return;
  int c = (int)(i / HW), hw = (int)(i % HW);
  int h = hw / IMG_H, w = hw % IMG_H;
  unsigned q = idx[(h / 3) * 64 + (w / 3)];
  int sh = (int)(q >> 6) * 3 + (h % 3);
  int sw = (int)(q & 63) * 3 + (w % 3);
  float v = style[(size_t)c * HW + (size_t)sh * IMG_H + sw] * mask[hw];
  out[i] = __float2bfloat16(v);
}

// ---------------- stage 4: Gram = X @ X^T, deterministic split-K -----------
__global__ void __launch_bounds__(256)
gemm_nt_store(const __hip_bfloat16* __restrict__ A,   // [M,K]
              const __hip_bfloat16* __restrict__ B,   // [N,K]
              float* __restrict__ Cout,               // [KSPLIT,M,N]
              int M, int N, int K, int kChunk) {
  __shared__ __hip_bfloat16 shA[2 * A_STAGE_ELEMS];
  __shared__ __hip_bfloat16 shB[2 * B_STAGE_ELEMS];
  int tilesN = N / TILE_N;
  int m0g = ((int)blockIdx.x / tilesN) * TILE_M;
  int n0g = ((int)blockIdx.x % tilesN) * TILE_N;
  int k0 = (int)blockIdx.y * kChunk;

  v8f acc[2][4] = {};
  gemm_tile_pipeline(A, B, K, k0, k0 + kChunk, m0g, n0g, shA, shB, acc);

  float* out = Cout + (size_t)blockIdx.y * M * N;
  int lane = (int)threadIdx.x & 31;
  int r = lane & 15, half = lane >> 4;
  int waveId = (int)threadIdx.x >> 5;
  int m0 = m0g + (waveId >> 2) * 32;
  int n0 = n0g + (waveId & 3) * 64;
  for (int ms = 0; ms < 2; ++ms)
    for (int ns = 0; ns < 4; ++ns)
      for (int j = 0; j < 8; ++j) {
        int row = m0 + ms * 16 + j + (half << 3);
        int col = n0 + ns * 16 + r;
        out[(size_t)row * N + col] = acc[ms][ns][j];
      }
}

// ---------------- stage 5: scalar reductions (single-block, deterministic) -
__global__ void mask_sum_kernel(const float* __restrict__ mask,
                                float* __restrict__ out) {
  __shared__ double sd[256];
  double s = 0.0;
  for (int i = threadIdx.x; i < HW; i += 256) s += (double)mask[i];
  sd[threadIdx.x] = s;
  __syncthreads();
  for (int st = 128; st > 0; st >>= 1) {
    if ((int)threadIdx.x < st) sd[threadIdx.x] += sd[threadIdx.x + st];
    __syncthreads();
  }
  if (threadIdx.x == 0) out[0] = (float)sd[0];
}

__global__ void loss_kernel(const float* __restrict__ GaP,  // [KSPLIT,512,512]
                            const float* __restrict__ GbP,
                            const float* __restrict__ msum,
                            float* __restrict__ out) {
  __shared__ double sd[256];
  const int NG = GRAM_N * GRAM_N;
  double s = 0.0;
  for (int i = threadIdx.x; i < NG; i += 256) {
    float ga = 0.f, gb = 0.f;
    for (int p = 0; p < KSPLIT; ++p) {
      ga += GaP[(size_t)p * NG + i];
      gb += GbP[(size_t)p * NG + i];
    }
    double d = (double)ga - (double)gb;
    s += d * d;
  }
  sd[threadIdx.x] = s;
  __syncthreads();
  for (int st = 128; st > 0; st >>= 1) {
    if ((int)threadIdx.x < st) sd[threadIdx.x] += sd[threadIdx.x + st];
    __syncthreads();
  }
  if (threadIdx.x == 0) {
    double ms = (double)msum[0] * (double)C_CH;      // mask expanded over C
    out[0] = (float)(s / (ms * ms) / (double)NG * 100.0);
  }
}

// ---------------------------------------------------------------------------
extern "C" void kernel_launch(void* const* d_in, const int* in_sizes, int n_in,
                              void* d_out, int out_size, void* d_ws, size_t ws_size,
                              hipStream_t stream) {
  (void)in_sizes; (void)n_in; (void)out_size; (void)ws_size;
  const float* input    = (const float*)d_in[0];   // [1,512,192,192]
  const float* style_fm = (const float*)d_in[1];
  const float* img_fm   = (const float*)d_in[2];
  const float* mask     = (const float*)d_in[3];   // [1,1,192,192]
  float* out = (float*)d_out;

  char* ws = (char*)d_ws;
  const size_t PATBYTES = (size_t)NPATCH * KPAT * 2;          // 37,748,736
  const size_t SP_OFF   = 0;                       // later reused for Xm
  const size_t IP_OFF   = PATBYTES;                // later reused for Xin
  const size_t PK_OFF   = 2 * PATBYTES;
  const size_t IDX_OFF  = PK_OFF + NPATCH * 8;
  const size_t GA_OFF   = IDX_OFF + NPATCH * 4;
  const size_t GB_OFF   = GA_OFF + (size_t)KSPLIT * GRAM_N * GRAM_N * 4;
  const size_t MS_OFF   = GB_OFF + (size_t)KSPLIT * GRAM_N * GRAM_N * 4;

  __hip_bfloat16* SP = (__hip_bfloat16*)(ws + SP_OFF);
  __hip_bfloat16* IP = (__hip_bfloat16*)(ws + IP_OFF);
  unsigned long long* packed = (unsigned long long*)(ws + PK_OFF);
  unsigned* idx = (unsigned*)(ws + IDX_OFF);
  __hip_bfloat16* Xm  = (__hip_bfloat16*)(ws + SP_OFF);   // reuse after argmax
  __hip_bfloat16* Xin = (__hip_bfloat16*)(ws + IP_OFF);   // reuse after argmax
  float* GaP = (float*)(ws + GA_OFF);   // style gram partials (matched)
  float* GbP = (float*)(ws + GB_OFF);   // img gram partials (input)
  float* msp = (float*)(ws + MS_OFF);

  const int PACK_BLOCKS = (int)(((size_t)NPATCH * KPAT + 255) / 256);   // 73728

  // 1) bf16 patch matrices
  pack_patches<<<PACK_BLOCKS, 256, 0, stream>>>(style_fm, SP);
  pack_patches<<<PACK_BLOCKS, 256, 0, stream>>>(img_fm, IP);

  // 2) scores GEMM + fused argmax
  hipMemsetAsync(packed, 0, NPATCH * sizeof(unsigned long long), stream);
  {
    int blocks = (NPATCH / TILE_M) * (NPATCH / TILE_N);   // 64*16 = 1024
    gemm_nt_argmax<<<blocks, 256, 0, stream>>>(IP, SP, packed,
                                               NPATCH, NPATCH, KPAT);
  }
  decode_idx<<<(NPATCH + 255) / 256, 256, 0, stream>>>(packed, idx);

  // 3) masked Gram operands (SP/IP regions reused)
  pack_masked<<<PACK_BLOCKS, 256, 0, stream>>>(input, mask, Xin);
  pack_matched<<<PACK_BLOCKS, 256, 0, stream>>>(style_fm, mask, idx, Xm);

  // 4) Gram GEMMs with split-K partials
  {
    dim3 grid((GRAM_N / TILE_M) * (GRAM_N / TILE_N), KSPLIT);   // (16, 8)
    gemm_nt_store<<<grid, 256, 0, stream>>>(Xm, Xm, GaP, GRAM_N, GRAM_N, HW,
                                            GRAM_KCHUNK);
    gemm_nt_store<<<grid, 256, 0, stream>>>(Xin, Xin, GbP, GRAM_N, GRAM_N, HW,
                                            GRAM_KCHUNK);
  }

  // 5) loss
  mask_sum_kernel<<<1, 256, 0, stream>>>(mask, msp);
  loss_kernel<<<1, 256, 0, stream>>>(GaP, GbP, msp, out);
}
